// VisnetBlock_62440234549684
// MI455X (gfx1250) — compile-verified, matrix operand order checked by reading, not run on hardware
//
#include <hip/hip_runtime.h>

// ---------------------------------------------------------------------------
// VisNet block for gfx1250 (MI455X). bf16 WMMA (v_wmma_f32_16x16x32_bf16) for
// all dense GEMMs; graph gather/scatter via f32 global atomics.
// GEMM: 8 waves/block, each wave = 2 M-tiles x 4 N-tiles (B-fragment reuse,
// 1.5 b128 loads per WMMA).
// ---------------------------------------------------------------------------

typedef __attribute__((ext_vector_type(16))) __bf16 v16bf;
typedef __attribute__((ext_vector_type(8)))  float  v8f;
typedef __bf16 bf16;

#define NNODE 10000
#define NEDGE 160000
#define CH    256
#define NH    8
#define KD    256   // GEMM K dimension (always CH)

__device__ __forceinline__ float silu_f(float x) {
    return x / (1.0f + __expf(-x));
}

__device__ __forceinline__ float wave_sum32(float v) {
    #pragma unroll
    for (int off = 16; off > 0; off >>= 1) v += __shfl_xor(v, off, 32);
    return v;
}

// Load one 16x32 bf16 WMMA fragment row (A layout; B uses same pattern on Wt rows).
// base: dword pointer to this lane's row of 256 bf16 (128 dwords).
// kb:   dword index of K-chunk start for this lane-half.
__device__ __forceinline__ v16bf load_frag(const unsigned int* __restrict__ base, int kb) {
    union { v16bf v; unsigned int u[8]; } t;
    #pragma unroll
    for (int r = 0; r < 4; ++r) t.u[r] = base[kb + r];          // K = 2r(+half)
    #pragma unroll
    for (int r = 4; r < 8; ++r) t.u[r] = base[kb + 4 + r];      // K = 16+2(r-4)(+half)
    return t.v;
}

// ---------------------------------------------------------------------------
// Generic bf16 WMMA GEMM: out[M,Nn] = act(A[M,256] @ Wt[Nn,256]^T + bias)
// Wt stores the weight TRANSPOSED (row n = output column n, K contiguous).
// Each wave: 2 consecutive 16-row M-tiles x 64 cols (8 WMMA accumulators);
// the 4 B fragments per K-step are shared by both M-tiles.
// ---------------------------------------------------------------------------
template<int ACT, int OUTBF>
__global__ __launch_bounds__(256) void gemm_bf16_wmma(
    const bf16* __restrict__ A, const bf16* __restrict__ Wt,
    const float* __restrict__ bias, void* __restrict__ out,
    int M, int Nn)
{
    const int lane   = threadIdx.x & 31;
    const int wave   = threadIdx.x >> 5;
    const int mtile0 = (blockIdx.x * 16 + wave * 2) * 16;
    if (mtile0 >= M) return;                      // wave-uniform exit
    const int mtile1 = mtile0 + 16;
    const bool has1  = (mtile1 < M);              // wave-uniform
    const int ntile  = blockIdx.y * 64;
    const int l15    = lane & 15;
    const int hi     = lane >> 4;

    const unsigned int* Arow0 = (const unsigned int*)(A + (size_t)(mtile0 + l15) * KD);
    const unsigned int* Arow1 = (const unsigned int*)(A + (size_t)((has1 ? mtile1 : mtile0) + l15) * KD);
    const unsigned int* Brow  = (const unsigned int*)(Wt + (size_t)(ntile + l15) * KD);
    const int bstride = 16 * (KD / 2);            // 16 Wt rows in dwords

    v8f acc[8] = {};                              // [mt*4 + j]
    #pragma unroll
    for (int kk = 0; kk < KD; kk += 32) {
        const int kb = (kk + hi * 8) >> 1;
        v16bf a0 = load_frag(Arow0, kb);
        v16bf a1 = load_frag(Arow1, kb);
        v16bf b0 = load_frag(Brow,               kb);
        v16bf b1 = load_frag(Brow +     bstride, kb);
        v16bf b2 = load_frag(Brow + 2 * bstride, kb);
        v16bf b3 = load_frag(Brow + 3 * bstride, kb);
        acc[0] = __builtin_amdgcn_wmma_f32_16x16x32_bf16(false, a0, false, b0, (short)0, acc[0], false, false);
        acc[4] = __builtin_amdgcn_wmma_f32_16x16x32_bf16(false, a1, false, b0, (short)0, acc[4], false, false);
        acc[1] = __builtin_amdgcn_wmma_f32_16x16x32_bf16(false, a0, false, b1, (short)0, acc[1], false, false);
        acc[5] = __builtin_amdgcn_wmma_f32_16x16x32_bf16(false, a1, false, b1, (short)0, acc[5], false, false);
        acc[2] = __builtin_amdgcn_wmma_f32_16x16x32_bf16(false, a0, false, b2, (short)0, acc[2], false, false);
        acc[6] = __builtin_amdgcn_wmma_f32_16x16x32_bf16(false, a1, false, b2, (short)0, acc[6], false, false);
        acc[3] = __builtin_amdgcn_wmma_f32_16x16x32_bf16(false, a0, false, b3, (short)0, acc[3], false, false);
        acc[7] = __builtin_amdgcn_wmma_f32_16x16x32_bf16(false, a1, false, b3, (short)0, acc[7], false, false);
    }

    // Epilogue. C/D layout: element r of acc = row (mtile + r + 8*hi), col ntile+j*16+l15.
    #pragma unroll
    for (int mt = 0; mt < 2; ++mt) {
        if (mt == 1 && !has1) break;              // wave-uniform
        const int mbase = mt ? mtile1 : mtile0;
        #pragma unroll
        for (int j = 0; j < 4; ++j) {
            const int n = ntile + j * 16 + l15;
            const float bv = bias ? bias[n] : 0.0f;
            #pragma unroll
            for (int r = 0; r < 8; ++r) {
                const int row = mbase + r + 8 * hi;
                float v = acc[mt * 4 + j][r] + bv;
                if (ACT) v = silu_f(v);
                const size_t idx = (size_t)row * Nn + n;
                if (OUTBF) ((bf16*)out)[idx] = (bf16)v;
                else       ((float*)out)[idx] = v;
            }
        }
    }
}

// ---------------------------------------------------------------------------
// Prep kernels
// ---------------------------------------------------------------------------

// LayerNorm over C=256, one wave per row, output bf16.
__global__ __launch_bounds__(256) void ln_kernel(
    const float* __restrict__ x, const float* __restrict__ sc,
    const float* __restrict__ bi, bf16* __restrict__ out, int rows)
{
    const int wave = threadIdx.x >> 5, lane = threadIdx.x & 31;
    const int row = blockIdx.x * 8 + wave;
    if (row >= rows) return;
    const float* xr = x + (size_t)row * CH;
    float vals[8], s = 0.f;
    #pragma unroll
    for (int i = 0; i < 8; ++i) { vals[i] = xr[lane + i * 32]; s += vals[i]; }
    const float mu = wave_sum32(s) * (1.0f / CH);
    float var = 0.f;
    #pragma unroll
    for (int i = 0; i < 8; ++i) { float d = vals[i] - mu; var += d * d; }
    const float rstd = rsqrtf(wave_sum32(var) * (1.0f / CH) + 1e-5f);
    #pragma unroll
    for (int i = 0; i < 8; ++i) {
        const int c = lane + i * 32;
        out[(size_t)row * CH + c] = (bf16)((vals[i] - mu) * rstd * sc[c] + bi[c]);
    }
}

// vec = vector_feats * vln_w  -> f32 + bf16 copies. One block per row of [3N, C].
__global__ __launch_bounds__(256) void vecscale_kernel(
    const float* __restrict__ vf, const float* __restrict__ w,
    float* __restrict__ outf, bf16* __restrict__ outb)
{
    const size_t i = (size_t)blockIdx.x * CH + threadIdx.x;
    const float v = vf[i] * w[threadIdx.x];
    outf[i] = v; outb[i] = (bf16)v;
}

__global__ __launch_bounds__(256) void f2b_kernel(const float* __restrict__ in,
                                                  bf16* __restrict__ out, size_t n)
{
    const size_t i = (size_t)blockIdx.x * 256 + threadIdx.x;
    if (i < n) out[i] = (bf16)in[i];
}

// Wt[n*K + k] = (bf16) W[k*N + n]
__global__ __launch_bounds__(256) void transpose_w_kernel(
    const float* __restrict__ W, bf16* __restrict__ Wt, int Kd, int Nd)
{
    const int idx = blockIdx.x * 256 + threadIdx.x;
    if (idx >= Kd * Nd) return;
    const int k = idx / Nd, n = idx % Nd;
    Wt[(size_t)n * Kd + k] = (bf16)W[(size_t)k * Nd + n];
}

__global__ void concat_bias_kernel(const float* a, const float* b, const float* c,
                                   float* __restrict__ out)
{
    const int i = threadIdx.x;
    out[i] = a[i]; out[CH + i] = b[i]; out[2 * CH + i] = c[i];
}

__global__ __launch_bounds__(256) void zero_kernel(float* p, size_t n) {
    const size_t i = (size_t)blockIdx.x * 256 + threadIdx.x;
    if (i < n) p[i] = 0.0f;
}

// ---------------------------------------------------------------------------
// Graph kernels
// ---------------------------------------------------------------------------

// attn + vmsg: one wave per (edge, head); lane = d (D == 32 == wave width).
__global__ __launch_bounds__(256) void edge_attn_kernel(
    const float* __restrict__ qkv, const bf16* __restrict__ eo,
    const float* __restrict__ dist,
    const int* __restrict__ snd, const int* __restrict__ rcv,
    bf16* __restrict__ vmsg)
{
    const int e = blockIdx.x;
    const int h = threadIdx.x >> 5, d = threadIdx.x & 31;
    const int r = rcv[e], s = snd[e];
    const int cd = h * 32 + d;
    const float qv  = qkv[(size_t)r * 768 + cd];
    const float kv  = qkv[(size_t)s * 768 + 256 + cd];
    const float vv  = qkv[(size_t)s * 768 + 512 + cd];
    const float dkv = (float)eo[(size_t)e * 768 + cd];
    const float dvv = (float)eo[(size_t)e * 768 + 256 + cd];
    const float dot = wave_sum32(qv * kv * dkv);
    const float di  = dist[e];
    const float cut = 0.5f * (__cosf(di * 0.6283185307f) + 1.0f) * (di < 5.0f ? 1.0f : 0.0f);
    const float a   = silu_f(dot) * cut;
    vmsg[(size_t)e * CH + cd] = (bf16)(vv * dvv * a);
}

// vec_msg + masked scatter of vmsg and vec_msg (segment_sum via f32 atomics).
__global__ __launch_bounds__(256) void edge_scatter_kernel(
    const bf16* __restrict__ s12, const bf16* __restrict__ vmsg,
    const float* __restrict__ vecs, const float* __restrict__ dij,
    const float* __restrict__ mask,
    const int* __restrict__ snd, const int* __restrict__ rcv,
    float* __restrict__ xagg, float* __restrict__ vecout)
{
    const int e = blockIdx.x, c = threadIdx.x;
    const int s = snd[e], r = rcv[e];
    const float mk = mask[e];
    const float s1 = (float)s12[(size_t)e * 512 + c];
    const float s2 = (float)s12[(size_t)e * 512 + 256 + c];
    const float vm = (float)vmsg[(size_t)e * CH + c] * mk;
    __hip_atomic_fetch_add(&xagg[(size_t)r * CH + c], vm,
                           __ATOMIC_RELAXED, __HIP_MEMORY_SCOPE_AGENT);
    #pragma unroll
    for (int a = 0; a < 3; ++a) {
        const float dv = dij[(size_t)e * 3 + a];
        const float vj = vecs[((size_t)s * 3 + a) * CH + c];
        __hip_atomic_fetch_add(&vecout[((size_t)r * 3 + a) * CH + c],
                               (vj * s1 + s2 * dv) * mk,
                               __ATOMIC_RELAXED, __HIP_MEMORY_SCOPE_AGENT);
    }
}

// dx, dvec. pvall row layout per (node, axis): [0:256)=vec1 [256:512)=vec2
// [512:768)=vec3 [768:1024)=vec@Wtrg [1024:1280)=vec@Wsrc
__global__ __launch_bounds__(256) void final_node_kernel(
    const float* __restrict__ pvall, const float* __restrict__ o123,
    const float* __restrict__ vecout,
    float* __restrict__ dx, float* __restrict__ dvec)
{
    const int n = blockIdx.x, c = threadIdx.x;
    const float* pvb = pvall + (size_t)n * 3 * 1280;
    float vdot = 0.f;
    #pragma unroll
    for (int a = 0; a < 3; ++a) vdot += pvb[a * 1280 + c] * pvb[a * 1280 + 256 + c];
    const float o1 = o123[(size_t)n * 768 + c];
    const float o2 = o123[(size_t)n * 768 + 256 + c];
    const float o3 = o123[(size_t)n * 768 + 512 + c];
    dx[(size_t)n * CH + c] = vdot * o2 + o3;
    #pragma unroll
    for (int a = 0; a < 3; ++a)
        dvec[((size_t)n * 3 + a) * CH + c] =
            pvb[a * 1280 + 512 + c] * o1 + vecout[((size_t)n * 3 + a) * CH + c];
}

// df_ij via vector rejection of (vec@Wtrg)[recv], (vec@Wsrc)[send].
__global__ __launch_bounds__(256) void final_edge_kernel(
    const float* __restrict__ pvall, const bf16* __restrict__ eo,
    const float* __restrict__ dij, const float* __restrict__ mask,
    const int* __restrict__ snd, const int* __restrict__ rcv,
    float* __restrict__ dfout)
{
    const int e = blockIdx.x, c = threadIdx.x;
    const int r = rcv[e], s = snd[e];
    float dd[3], vt[3], vs[3];
    #pragma unroll
    for (int a = 0; a < 3; ++a) {
        dd[a] = dij[(size_t)e * 3 + a];
        vt[a] = pvall[((size_t)r * 3 + a) * 1280 + 768  + c];
        vs[a] = pvall[((size_t)s * 3 + a) * 1280 + 1024 + c];
    }
    float p1 = 0.f, p2 = 0.f;
    #pragma unroll
    for (int a = 0; a < 3; ++a) { p1 += vt[a] * dd[a]; p2 -= vs[a] * dd[a]; }
    float wdot = 0.f;
    #pragma unroll
    for (int a = 0; a < 3; ++a)
        wdot += (vt[a] - p1 * dd[a]) * (vs[a] + p2 * dd[a]);
    const float f = (float)eo[(size_t)e * 768 + 512 + c];
    dfout[(size_t)e * CH + c] = f * wdot * mask[e];
}

// ---------------------------------------------------------------------------
// Launch
// ---------------------------------------------------------------------------
extern "C" void kernel_launch(void* const* d_in, const int* in_sizes, int n_in,
                              void* d_out, int out_size, void* d_ws, size_t ws_size,
                              hipStream_t stream)
{
    const float* node_feats = (const float*)d_in[0];
    const float* edge_feats = (const float*)d_in[1];
    const float* vector_feats = (const float*)d_in[2];
    const float* distances = (const float*)d_in[3];
    const float* d_ij = (const float*)d_in[4];
    const float* mask = (const float*)d_in[5];
    const int*   senders = (const int*)d_in[6];
    const int*   receivers = (const int*)d_in[7];
    const float* ln_scale = (const float*)d_in[8];
    const float* ln_bias = (const float*)d_in[9];
    const float* vln_w = (const float*)d_in[10];
    const float* Wq = (const float*)d_in[11]; const float* bq = (const float*)d_in[12];
    const float* Wk = (const float*)d_in[13]; const float* bk = (const float*)d_in[14];
    const float* Wv = (const float*)d_in[15]; const float* bv = (const float*)d_in[16];
    const float* Wdk = (const float*)d_in[17]; const float* bdk = (const float*)d_in[18];
    const float* Wdv = (const float*)d_in[19]; const float* bdv = (const float*)d_in[20];
    const float* Wvec = (const float*)d_in[21];
    const float* Ws = (const float*)d_in[22]; const float* bs = (const float*)d_in[23];
    const float* Wo = (const float*)d_in[24]; const float* bo = (const float*)d_in[25];
    const float* Wf = (const float*)d_in[26]; const float* bf = (const float*)d_in[27];
    const float* Wsrc = (const float*)d_in[28];
    const float* Wtrg = (const float*)d_in[29];

    // --- workspace carve-up -------------------------------------------------
    char* wp = (char*)d_ws;
    auto carve = [&](size_t bytes) { void* p = (void*)wp; wp += (bytes + 255) & ~(size_t)255; return p; };
    bf16*  xb     = (bf16*) carve((size_t)NNODE * CH * 2);
    float* vecsF  = (float*)carve((size_t)NNODE * 3 * CH * 4);
    bf16*  vecsB  = (bf16*) carve((size_t)NNODE * 3 * CH * 2);
    bf16*  edgeB  = (bf16*) carve((size_t)NEDGE * CH * 2);
    bf16*  WtQKV  = (bf16*) carve((size_t)768  * KD * 2);
    bf16*  WtEdge = (bf16*) carve((size_t)768  * KD * 2);
    bf16*  WtVec  = (bf16*) carve((size_t)1280 * KD * 2);
    bf16*  WtS    = (bf16*) carve((size_t)512  * KD * 2);
    bf16*  WtO    = (bf16*) carve((size_t)768  * KD * 2);
    float* bQKV   = (float*)carve(768 * 4);
    float* bEdge  = (float*)carve(768 * 4);
    float* qkv    = (float*)carve((size_t)NNODE * 768 * 4);
    bf16*  edgeO  = (bf16*) carve((size_t)NEDGE * 768 * 2);
    float* pvall  = (float*)carve((size_t)NNODE * 3 * 1280 * 4);
    bf16*  vmsgB  = (bf16*) carve((size_t)NEDGE * CH * 2);
    bf16*  s12B   = (bf16*) carve((size_t)NEDGE * 512 * 2);
    float* xagg   = (float*)carve((size_t)NNODE * CH * 4);
    bf16*  xaggB  = (bf16*) carve((size_t)NNODE * CH * 2);
    float* vecout = (float*)carve((size_t)NNODE * 3 * CH * 4);
    float* o123   = (float*)carve((size_t)NNODE * 768 * 4);
    (void)ws_size; (void)n_in; (void)in_sizes; (void)out_size;

    float* out_dx   = (float*)d_out;                                   // [N,C]
    float* out_df   = out_dx + (size_t)NNODE * CH;                     // [E,C]
    float* out_dvec = out_df + (size_t)NEDGE * CH;                     // [N,3,C]

    const int WELEM = (CH * CH + 255) / 256;       // 256x256 transpose grid
    const int WELEM3 = (CH * 768 + 255) / 256;     // 256x768

    // GEMM grid helper: 16 M-tiles (256 rows) per block.
    auto gx = [](int M) { return (M / 16 + 15) / 16 + ((M % 256) ? 0 : 0); };

    // --- stage 0: conversions / transposes ----------------------------------
    transpose_w_kernel<<<WELEM, 256, 0, stream>>>(Wq,  WtQKV,            KD, CH);
    transpose_w_kernel<<<WELEM, 256, 0, stream>>>(Wk,  WtQKV + 256*KD,   KD, CH);
    transpose_w_kernel<<<WELEM, 256, 0, stream>>>(Wv,  WtQKV + 512*KD,   KD, CH);
    transpose_w_kernel<<<WELEM, 256, 0, stream>>>(Wdk, WtEdge,           KD, CH);
    transpose_w_kernel<<<WELEM, 256, 0, stream>>>(Wdv, WtEdge + 256*KD,  KD, CH);
    transpose_w_kernel<<<WELEM, 256, 0, stream>>>(Wf,  WtEdge + 512*KD,  KD, CH);
    transpose_w_kernel<<<WELEM3,256, 0, stream>>>(Wvec, WtVec,           KD, 768);
    transpose_w_kernel<<<WELEM, 256, 0, stream>>>(Wtrg, WtVec + 768*KD,  KD, CH);
    transpose_w_kernel<<<WELEM, 256, 0, stream>>>(Wsrc, WtVec + 1024*KD, KD, CH);
    transpose_w_kernel<<<(CH*512+255)/256, 256, 0, stream>>>(Ws, WtS,    KD, 512);
    transpose_w_kernel<<<WELEM3,256, 0, stream>>>(Wo,  WtO,              KD, 768);
    concat_bias_kernel<<<1, 256, 0, stream>>>(bq, bk, bv, bQKV);
    concat_bias_kernel<<<1, 256, 0, stream>>>(bdk, bdv, bf, bEdge);

    ln_kernel<<<(NNODE + 7) / 8, 256, 0, stream>>>(node_feats, ln_scale, ln_bias, xb, NNODE);
    vecscale_kernel<<<NNODE * 3, 256, 0, stream>>>(vector_feats, vln_w, vecsF, vecsB);
    f2b_kernel<<<(int)(((size_t)NEDGE * CH + 255) / 256), 256, 0, stream>>>(
        edge_feats, edgeB, (size_t)NEDGE * CH);
    zero_kernel<<<(int)(((size_t)NNODE * CH + 255) / 256), 256, 0, stream>>>(
        xagg, (size_t)NNODE * CH);
    zero_kernel<<<(int)(((size_t)NNODE * 3 * CH + 255) / 256), 256, 0, stream>>>(
        vecout, (size_t)NNODE * 3 * CH);

    // --- stage 1: independent GEMMs -----------------------------------------
    // qkv = x @ [Wq|Wk|Wv] + b  (f32 out)
    gemm_bf16_wmma<0, 0><<<dim3(gx(NNODE), 768 / 64), 256, 0, stream>>>(
        xb, WtQKV, bQKV, qkv, NNODE, 768);
    // dk|dv|f = silu(edge @ [Wdk|Wdv|Wf] + b)  (bf16 out)
    gemm_bf16_wmma<1, 1><<<dim3(gx(NEDGE), 768 / 64), 256, 0, stream>>>(
        edgeB, WtEdge, bEdge, edgeO, NEDGE, 768);
    // pv|vt|vs = vec @ [Wvec|Wtrg|Wsrc]  (f32 out, no bias)
    gemm_bf16_wmma<0, 0><<<dim3(gx(NNODE * 3), 1280 / 64), 256, 0, stream>>>(
        vecsB, WtVec, nullptr, pvall, NNODE * 3, 1280);

    // --- stage 2: attention message ------------------------------------------
    edge_attn_kernel<<<NEDGE, 256, 0, stream>>>(qkv, edgeO, distances,
                                                senders, receivers, vmsgB);

    // --- stage 3: s12 = silu(vmsg @ Ws + bs)  (bf16 out) ---------------------
    gemm_bf16_wmma<1, 1><<<dim3(gx(NEDGE), 512 / 64), 256, 0, stream>>>(
        vmsgB, WtS, bs, s12B, NEDGE, 512);

    // --- stage 4: masked scatter (segment sums) ------------------------------
    edge_scatter_kernel<<<NEDGE, 256, 0, stream>>>(s12B, vmsgB, vecsF, d_ij, mask,
                                                   senders, receivers, xagg, vecout);
    f2b_kernel<<<(int)(((size_t)NNODE * CH + 255) / 256), 256, 0, stream>>>(
        xagg, xaggB, (size_t)NNODE * CH);

    // --- stage 5: o123 = x_agg @ Wo + bo  (f32 out) --------------------------
    gemm_bf16_wmma<0, 0><<<dim3(gx(NNODE), 768 / 64), 256, 0, stream>>>(
        xaggB, WtO, bo, o123, NNODE, 768);

    // --- stage 6: outputs -----------------------------------------------------
    final_node_kernel<<<NNODE, 256, 0, stream>>>(pvall, o123, vecout, out_dx, out_dvec);
    final_edge_kernel<<<NEDGE, 256, 0, stream>>>(pvall, edgeO, d_ij, mask,
                                                 senders, receivers, out_df);
}
